// Qnet_12764642804108
// MI455X (gfx1250) — compile-verified, matrix operand order checked by reading, not run on hardware
//
#include <hip/hip_runtime.h>

// Problem constants (from reference)
constexpr int kB = 64;    // batch
constexpr int kW = 2048;  // timesteps
constexpr int kF = 64;    // input features
constexpr int kH = 128;   // hidden
constexpr int kZ = 32;    // latent
constexpr int kD = 128;   // dense
constexpr int kK = 4;     // planar flows
constexpr int kG = 3 * kH;  // 384 gates

typedef __attribute__((ext_vector_type(16))) _Float16 v16h;
typedef __attribute__((ext_vector_type(8)))  float    v8f;

__device__ __forceinline__ v8f wmma_f16(v16h a, v16h b, v8f c) {
  // D = A(16x32 f16) * B(32x16 f16) + C(16x16 f32)
  return __builtin_amdgcn_wmma_f32_16x16x32_f16(false, a, false, b, (short)0, c,
                                                false, false);
}

// A-fragment (16 rows x 32 K) from row-major f16 LDS matrix, row stride ld (halves).
__device__ __forceinline__ v16h frag_a(const _Float16* s, int ld, int col0, int lane) {
  const _Float16* p = s + (lane & 15) * ld + col0 + ((lane >> 4) << 3);
  v16h a;
#pragma unroll
  for (int e = 0; e < 8; ++e) a[e] = p[e];
#pragma unroll
  for (int e = 0; e < 8; ++e) a[8 + e] = p[16 + e];
  return a;
}

// Same, but source matrix is f32 in LDS (converted on the fly).
__device__ __forceinline__ v16h frag_a_f32(const float* s, int ld, int col0, int lane) {
  const float* p = s + (lane & 15) * ld + col0 + ((lane >> 4) << 3);
  v16h a;
#pragma unroll
  for (int e = 0; e < 8; ++e) a[e] = (_Float16)p[e];
#pragma unroll
  for (int e = 0; e < 8; ++e) a[8 + e] = (_Float16)p[16 + e];
  return a;
}

// B-fragment from pre-swizzled tile: 32 lanes x 16 halves, contiguous per lane.
__device__ __forceinline__ v16h frag_b(const _Float16* tile, int lane) {
  const _Float16* p = tile + (lane << 4);
  v16h b;
#pragma unroll
  for (int e = 0; e < 16; ++e) b[e] = p[e];
  return b;
}

// Pre-swizzle an (N x Kdim) row-major f32 weight (out,in layout == B^T) into
// 32x16 B tiles, tile linear index = nt*Kt + kt, 512 halves per tile.
__device__ void pack_b_tiles(_Float16* dst, const float* Wsrc, int N, int Kdim,
                             int tid, int nthr) {
  const int Kt = Kdim >> 5;
  const int total = (N >> 4) * Kt * 512;
  for (int idx = tid; idx < total; idx += nthr) {
    int t = idx >> 9, r = idx & 511;
    int ln = r >> 4, e = r & 15;
    int nt = t / Kt, kt = t - nt * Kt;
    int n = (nt << 4) + (ln & 15);
    int k = (kt << 5) + ((ln >> 4) << 3) + ((e >> 3) << 4) + (e & 7);
    dst[idx] = (_Float16)Wsrc[n * Kdim + k];
  }
}

// Fast transcendentals: raw v_rcp_f32 / v_exp_f32 / v_log_f32, no IEEE-div fixups.
__device__ __forceinline__ float fast_rcp(float v) { return __builtin_amdgcn_rcpf(v); }
__device__ __forceinline__ float sigmoidf_(float v) {
  return fast_rcp(1.0f + __expf(-v));           // exp(inf)->inf, rcp(inf)->0: safe
}
__device__ __forceinline__ float tanhf_(float v) {
  float xc = fminf(fmaxf(v, -15.0f), 15.0f);    // saturate to avoid inf*0
  float t = __expf(2.0f * xc);
  return (t - 1.0f) * fast_rcp(t + 1.0f);
}
__device__ __forceinline__ float softplusf_(float v) {
  return (v > 20.0f) ? v : __logf(1.0f + __expf(v));
}

// C/D fragment-major index: element (row m, col-in-tile n16) of a 16x16 f32 tile
// stored as [lane][8] contiguous floats (lane = n16 + 16*(m/8), slot = m%8).
__device__ __forceinline__ int cidx(int n16, int m) {
  return ((n16 + ((m >> 3) << 4)) << 3) + (m & 7);
}

// ---------------- Tensor Data Mover (gfx1250 TDM) ----------------
#if __has_builtin(__builtin_amdgcn_tensor_load_to_lds)
#define USE_TDM 1
typedef unsigned int u32x4 __attribute__((ext_vector_type(4)));
typedef int i32x4 __attribute__((ext_vector_type(4)));
typedef int i32x8 __attribute__((ext_vector_type(8)));

// 2-D f32 tile DMA: global (tile_h rows x tile_w elems, row stride in elems) -> LDS.
// D# bitfields per CDNA5 ISA (count=1, type=2=image,
// data_size code 2 = 4B, no multicast / pad / iterate; groups 2-4 zero: 2-D tile).
__device__ __forceinline__ void tdm_load_2d_f32(unsigned ldsOff, const float* gsrc,
                                                unsigned tile_w, unsigned tile_h,
                                                unsigned long long stride0) {
  unsigned long long ga = (unsigned long long)(uintptr_t)gsrc;
  u32x4 g0;
  g0.x = 1u;                                                // count=1, user mode
  g0.y = ldsOff;                                            // lds_addr (bytes)
  g0.z = (unsigned)ga;                                      // global_addr[31:0]
  g0.w = (unsigned)((ga >> 32) & 0x01ffffffu) | (2u << 30); // addr[56:32] | type=2
  i32x8 g1;
  g1[0] = (int)(2u << 16);                         // wg_mask=0, data_size=4B
  g1[1] = (int)((tile_w & 0xffffu) << 16);         // tensor_dim0[15:0] (== tile_w)
  g1[2] = (int)((tile_h & 0xffffu) << 16);         // tensor_dim0 hi=0 | tensor_dim1 lo
  g1[3] = (int)((tile_w & 0xffffu) << 16);         // tensor_dim1 hi=0 | tile_dim0
  g1[4] = (int)(tile_h & 0xffffu);                 // tile_dim1 | tile_dim2=0
  g1[5] = (int)(unsigned)stride0;                  // tensor_dim0_stride[31:0]
  g1[6] = (int)((stride0 >> 32) & 0xffffu);        // stride hi | dim1_stride lo=0
  g1[7] = 0;
  i32x4 zz4 = {0, 0, 0, 0};
  i32x8 zz8 = {0, 0, 0, 0, 0, 0, 0, 0};
  __builtin_amdgcn_tensor_load_to_lds(g0, g1, zz4, zz4, zz8, 0);
}
#endif

// Dynamic-LDS footprint (bytes)
constexpr int SMEM_BYTES = 282896;

extern "C" __global__ __launch_bounds__(256)
void vrnn_fused(const float* __restrict__ x, const float* __restrict__ noise,
                const float* __restrict__ W_ih, const float* __restrict__ W_hh,
                const float* __restrict__ b_ih, const float* __restrict__ b_hh,
                const float* __restrict__ Wd,  const float* __restrict__ bd,
                const float* __restrict__ Wmu, const float* __restrict__ bmu,
                const float* __restrict__ Wsig, const float* __restrict__ bsig,
                const float* __restrict__ pnf_u, const float* __restrict__ pnf_W,
                const float* __restrict__ pnf_b, float* __restrict__ out) {
  extern __shared__ char smem[];
  // f16 region (halves)
  _Float16* WhhB = (_Float16*)smem;            // 24*4*512 = 49152
  _Float16* WihB = WhhB + 49152;               // 24*2*512 = 24576
  _Float16* WdB  = WihB + 24576;               //  8*5*512 = 20480
  _Float16* WmsB = WdB  + 20480;               //  4*4*512 =  8192
  _Float16* WfB  = WmsB + 8192;                //  4*2*512 =  4096
  _Float16* h16  = WfB  + 4096;                // 16*128
  _Float16* in2  = h16  + 2048;                // 16*160  [z | h]
  _Float16* z16  = in2  + 2560;                // 16*32
  _Float16* d16  = z16  + 512;                 // 16*128
  // f32 region
  float* xbf   = (float*)(d16 + 2048);         // 2 x 16*64  (double buffer, f32)
  float* nbuf  = xbf   + 2048;                 // 2 x 16*32  (double buffer)
  float* bufru = nbuf  + 1024;                 // 16 tiles x 256  (r|u fused pre-acts)
  float* bufxn = bufru + 4096;                 //  8 tiles x 256  (xg, n-gate)
  float* bufgn = bufxn + 2048;                 //  8 tiles x 256  (gh, n-gate)
  float* musig = bufgn + 2048;                 //  4 tiles x 256  ([mu|sig] frag-major)
  float* zf32  = musig + 1024;                 // 16*32
  float* sbih  = zf32  + 512;                  // 384
  float* sbhh  = sbih  + 384;                  // 384
  float* sbd   = sbhh  + 384;                  // 128
  float* sbms  = sbd   + 128;                  // 64 [bmu | bsig]
  float* spfb  = sbms  + 64;                   // 4*32
  float* spfu  = spfb  + 128;                  // 4

  const int tid  = threadIdx.x;
  const int lane = tid & 31;
  const int wv   = tid >> 5;
  const int b0   = blockIdx.x << 4;  // 16 batch rows per workgroup

#ifdef USE_TDM
#if __has_builtin(__builtin_amdgcn_groupstaticsize)
  const unsigned ldsBase = __builtin_amdgcn_groupstaticsize();
#else
  const unsigned ldsBase = 0u;
#endif
#endif

  // ---- One-time: pre-swizzle all weights into WMMA B-tile layout (f16) ----
  pack_b_tiles(WhhB, W_hh, kG, kH, tid, 256);
  pack_b_tiles(WihB, W_ih, kG, kF, tid, 256);
  pack_b_tiles(WdB,  Wd,   kD, kZ + kH, tid, 256);
  for (int idx = tid; idx < 16 * 512; idx += 256) {   // [Wmu ; Wsig] fused, N=64,K=128
    int t = idx >> 9, r = idx & 511, ln = r >> 4, e = r & 15;
    int nt = t >> 2, kt = t & 3;
    int n = (nt << 4) + (ln & 15);
    int k = (kt << 5) + ((ln >> 4) << 3) + ((e >> 3) << 4) + (e & 7);
    float v = (n < kZ) ? Wmu[n * kD + k] : Wsig[(n - kZ) * kD + k];
    WmsB[idx] = (_Float16)v;
  }
  for (int kf = 0; kf < kK; ++kf)
    pack_b_tiles(WfB + kf * 1024, pnf_W + kf * kZ * kZ, kZ, kZ, tid, 256);

  for (int i = tid; i < kG; i += 256) { sbih[i] = b_ih[i]; sbhh[i] = b_hh[i]; }
  if (tid < kD) sbd[tid] = bd[tid];
  if (tid < 64) sbms[tid] = (tid < kZ) ? bmu[tid] : bsig[tid - kZ];
  if (tid < kK * kZ) spfb[tid] = pnf_b[tid];
  if (tid < kK) spfu[tid] = pnf_u[tid];

  for (int i = tid; i < 16 * kH; i += 256) h16[i] = (_Float16)0.f;
  for (int i = tid; i < 16 * kZ; i += 256) z16[i] = (_Float16)0.f;

  // ---- First timestep inputs ----
#ifdef USE_TDM
  if (wv == 0) {
    tdm_load_2d_f32(ldsBase + (unsigned)((char*)xbf - smem),
                    &x[(size_t)b0 * kW * kF], kF, 16, (unsigned long long)kW * kF);
    tdm_load_2d_f32(ldsBase + (unsigned)((char*)nbuf - smem),
                    &noise[(size_t)b0 * kW * kZ], kZ, 16, (unsigned long long)kW * kZ);
    __builtin_amdgcn_s_wait_tensorcnt((short)0);
  }
#else
#pragma unroll
  for (int i = 0; i < 4; ++i) {
    int idx = tid + (i << 8); int m = idx >> 6, f = idx & 63;
    xbf[idx] = x[((size_t)(b0 + m) * kW + 0) * kF + f];
  }
#pragma unroll
  for (int i = 0; i < 2; ++i) {
    int idx = tid + (i << 8); int m = idx >> 5, zc = idx & 31;
    nbuf[idx] = noise[((size_t)(b0 + m) * kW + 0) * kZ + zc];
  }
#endif
  __syncthreads();

  float* out_z  = out;
  float* out_mu = out + (size_t)kB * kW * kZ;
  float* out_lv = out_mu + (size_t)kB * kW * kZ;

  for (int w = 0; w < kW; ++w) {
    const int cur = w & 1, nxt = cur ^ 1;

    // ---- Kick off next step's input staging (overlaps the WMMA phases) ----
#ifdef USE_TDM
    if (w + 1 < kW && wv == 0) {
      tdm_load_2d_f32(ldsBase + (unsigned)((char*)(xbf + nxt * 1024) - smem),
                      &x[((size_t)b0 * kW + (w + 1)) * kF], kF, 16,
                      (unsigned long long)kW * kF);
      tdm_load_2d_f32(ldsBase + (unsigned)((char*)(nbuf + nxt * 512) - smem),
                      &noise[((size_t)b0 * kW + (w + 1)) * kZ], kZ, 16,
                      (unsigned long long)kW * kZ);
    }
#else
    float px[4]; float pn[2];
    if (w + 1 < kW) {
#pragma unroll
      for (int i = 0; i < 4; ++i) {
        int idx = tid + (i << 8); int m = idx >> 6, f = idx & 63;
        px[i] = x[((size_t)(b0 + m) * kW + (w + 1)) * kF + f];
      }
#pragma unroll
      for (int i = 0; i < 2; ++i) {
        int idx = tid + (i << 8); int m = idx >> 5, zc = idx & 31;
        pn[i] = noise[((size_t)(b0 + m) * kW + (w + 1)) * kZ + zc];
      }
    }
#endif
    if (w + 8 < kW) {  // deep stream prefetch -> global_prefetch_b8 (L2 warm)
      int m = tid >> 6, f = tid & 63;
      __builtin_prefetch(&x[((size_t)(b0 + m) * kW + (w + 8)) * kF + f], 0, 1);
    }

    // ---- Phase A: gate projections.  Tiles 0..15 (r,u): xg+gh fused into one
    //      accumulator; tiles 16..23 (n): xg and gh kept separate. ----
    {
      const float* xcur = xbf + cur * 1024;
      v16h ax0 = frag_a_f32(xcur, kF, 0, lane);
      v16h ax1 = frag_a_f32(xcur, kF, 32, lane);
      v16h ah[4];
#pragma unroll
      for (int kt = 0; kt < 4; ++kt) ah[kt] = frag_a(h16, kH, kt * 32, lane);
#pragma unroll
      for (int t = 0; t < 3; ++t) {
        const int nt = wv * 3 + t;
        if (nt < 16) {
          v8f acc = {0.f, 0.f, 0.f, 0.f, 0.f, 0.f, 0.f, 0.f};
          acc = wmma_f16(ax0, frag_b(WihB + (nt * 2 + 0) * 512, lane), acc);
          acc = wmma_f16(ax1, frag_b(WihB + (nt * 2 + 1) * 512, lane), acc);
#pragma unroll
          for (int kt = 0; kt < 4; ++kt)
            acc = wmma_f16(ah[kt], frag_b(WhhB + (nt * 4 + kt) * 512, lane), acc);
          float* dst = bufru + (nt << 8) + (lane << 3);  // lane-contiguous 32B
#pragma unroll
          for (int j = 0; j < 8; ++j) dst[j] = acc[j];
        } else {
          v8f accx = {0.f, 0.f, 0.f, 0.f, 0.f, 0.f, 0.f, 0.f};
          v8f acch = {0.f, 0.f, 0.f, 0.f, 0.f, 0.f, 0.f, 0.f};
          accx = wmma_f16(ax0, frag_b(WihB + (nt * 2 + 0) * 512, lane), accx);
          accx = wmma_f16(ax1, frag_b(WihB + (nt * 2 + 1) * 512, lane), accx);
#pragma unroll
          for (int kt = 0; kt < 4; ++kt)
            acch = wmma_f16(ah[kt], frag_b(WhhB + (nt * 4 + kt) * 512, lane), acch);
          float* dx = bufxn + ((nt - 16) << 8) + (lane << 3);
          float* dg = bufgn + ((nt - 16) << 8) + (lane << 3);
#pragma unroll
          for (int j = 0; j < 8; ++j) { dx[j] = accx[j]; dg[j] = acch[j]; }
        }
      }
    }
    __syncthreads();

    // ---- Phase B: GRU elementwise update, build dense input [z_t | h_new] ----
    for (int i = tid; i < 16 * kH; i += 256) {
      int m = i >> 7, c = i & 127;
      int fi = cidx(c & 15, m);
      int t0 = c >> 4;
      float r = sigmoidf_(bufru[(t0 << 8) + fi] + sbih[c] + sbhh[c]);
      float u = sigmoidf_(bufru[((8 + t0) << 8) + fi] + sbih[kH + c] + sbhh[kH + c]);
      float n = tanhf_(bufxn[(t0 << 8) + fi] + sbih[2 * kH + c] +
                       r * (bufgn[(t0 << 8) + fi] + sbhh[2 * kH + c]));
      float hprev = (float)h16[m * kH + c];
      float hv = (1.0f - u) * n + u * hprev;
      _Float16 hh = (_Float16)hv;
      h16[m * kH + c] = hh;
      in2[m * (kZ + kH) + kZ + c] = hh;
    }
    for (int i = tid; i < 16 * kZ; i += 256) {
      int m = i >> 5, zc = i & 31;
      in2[m * (kZ + kH) + zc] = z16[m * kZ + zc];  // z_t = carried (pre-update) z
    }
    __syncthreads();

    // ---- Phase C: dense = [z|h] @ Wd^T + bd   (K=160, N=128; 1 tile/wave) ----
    {
      v16h ad[5];
#pragma unroll
      for (int kt = 0; kt < 5; ++kt) ad[kt] = frag_a(in2, kZ + kH, kt * 32, lane);
      v8f acc = {0.f, 0.f, 0.f, 0.f, 0.f, 0.f, 0.f, 0.f};
#pragma unroll
      for (int kt = 0; kt < 5; ++kt)
        acc = wmma_f16(ad[kt], frag_b(WdB + (wv * 5 + kt) * 512, lane), acc);
      const int n = (wv << 4) + (lane & 15);
      const int mb = (lane >> 4) << 3;
      const float bv = sbd[n];
#pragma unroll
      for (int j = 0; j < 8; ++j) d16[(mb + j) * kD + n] = (_Float16)(acc[j] + bv);
    }
    __syncthreads();

    // ---- Phase D: [mu | sig_pre] = dense @ [Wmu;Wsig]^T  (K=128, N=64) ----
    if (wv < 4) {
      v16h ad[4];
#pragma unroll
      for (int kt = 0; kt < 4; ++kt) ad[kt] = frag_a(d16, kD, kt * 32, lane);
      v8f acc = {0.f, 0.f, 0.f, 0.f, 0.f, 0.f, 0.f, 0.f};
#pragma unroll
      for (int kt = 0; kt < 4; ++kt)
        acc = wmma_f16(ad[kt], frag_b(WmsB + (wv * 4 + kt) * 512, lane), acc);
      const float bv = sbms[(wv << 4) + (lane & 15)];
      float* dst = musig + (wv << 8) + (lane << 3);  // fragment-major, contiguous
#pragma unroll
      for (int j = 0; j < 8; ++j) dst[j] = acc[j] + bv;
    }
    __syncthreads();

    // ---- Phase E: sample z, emit mu/logvar ----
    for (int i = tid; i < 16 * kZ; i += 256) {
      int m = i >> 5, zc = i & 31;
      int fi = cidx(zc & 15, m);
      float mu = musig[((zc >> 4) << 8) + fi];
      float lv = softplusf_(musig[((2 + (zc >> 4)) << 8) + fi]);
      float zv = mu + __expf(0.5f * lv) * nbuf[cur * 512 + i];
      zf32[i] = zv;
      z16[i] = (_Float16)zv;
      size_t row = ((size_t)(b0 + m) * kW + w) * kZ + zc;
      out_mu[row] = mu;
      out_lv[row] = lv;
    }
    __syncthreads();

    // ---- Phase F: planar flows (wave 0 only; intra-wave DS ops are ordered,
    //      so the 4 dependent flow iterations need no workgroup barriers) ----
    if (wv == 0) {
      for (int kf = 0; kf < kK; ++kf) {
        v16h az = frag_a(z16, kZ, 0, lane);
        v8f a0 = {0.f, 0.f, 0.f, 0.f, 0.f, 0.f, 0.f, 0.f};
        v8f a1 = {0.f, 0.f, 0.f, 0.f, 0.f, 0.f, 0.f, 0.f};
        a0 = wmma_f16(az, frag_b(WfB + (kf * 2 + 0) * 512, lane), a0);
        a1 = wmma_f16(az, frag_b(WfB + (kf * 2 + 1) * 512, lane), a1);
        const int nn = lane & 15;
        const int mb = (lane >> 4) << 3;
        const float uk = spfu[kf];
#pragma unroll
        for (int j = 0; j < 8; ++j) {
          int m = mb + j;
          float z0 = zf32[m * kZ + nn] + uk * tanhf_(a0[j] + spfb[kf * kZ + nn]);
          float z1 = zf32[m * kZ + 16 + nn] + uk * tanhf_(a1[j] + spfb[kf * kZ + 16 + nn]);
          zf32[m * kZ + nn] = z0;       z16[m * kZ + nn] = (_Float16)z0;
          zf32[m * kZ + 16 + nn] = z1;  z16[m * kZ + 16 + nn] = (_Float16)z1;
        }
      }
      for (int i = lane; i < 16 * kZ; i += 32) {
        int m = i >> 5, zc = i & 31;
        out_z[((size_t)(b0 + m) * kW + w) * kZ + zc] = zf32[i];
      }
    }

    // ---- Phase G: make next-step inputs visible before the step boundary ----
#ifdef USE_TDM
    if (wv == 0) __builtin_amdgcn_s_wait_tensorcnt((short)0);
#else
    if (w + 1 < kW) {
#pragma unroll
      for (int i = 0; i < 4; ++i) xbf[nxt * 1024 + tid + (i << 8)] = px[i];
#pragma unroll
      for (int i = 0; i < 2; ++i) nbuf[nxt * 512 + tid + (i << 8)] = pn[i];
    }
#endif
    __syncthreads();
  }
}

extern "C" void kernel_launch(void* const* d_in, const int* in_sizes, int n_in,
                              void* d_out, int out_size, void* d_ws, size_t ws_size,
                              hipStream_t stream) {
  (void)in_sizes; (void)n_in; (void)out_size; (void)d_ws; (void)ws_size;
  const float* x     = (const float*)d_in[0];
  const float* noise = (const float*)d_in[1];
  const float* W_ih  = (const float*)d_in[2];
  const float* W_hh  = (const float*)d_in[3];
  const float* b_ih  = (const float*)d_in[4];
  const float* b_hh  = (const float*)d_in[5];
  const float* Wd    = (const float*)d_in[6];
  const float* bd    = (const float*)d_in[7];
  const float* Wmu   = (const float*)d_in[8];
  const float* bmu   = (const float*)d_in[9];
  const float* Wsig  = (const float*)d_in[10];
  const float* bsig  = (const float*)d_in[11];
  const float* pnf_u = (const float*)d_in[12];
  const float* pnf_W = (const float*)d_in[13];
  const float* pnf_b = (const float*)d_in[14];
  float* out = (float*)d_out;

  (void)hipFuncSetAttribute((const void*)vrnn_fused,
                            hipFuncAttributeMaxDynamicSharedMemorySize, SMEM_BYTES);
  vrnn_fused<<<dim3(kB / 16), dim3(256), SMEM_BYTES, stream>>>(
      x, noise, W_ih, W_hh, b_ih, b_hh, Wd, bd, Wmu, bmu, Wsig, bsig,
      pnf_u, pnf_W, pnf_b, out);
}
